// VKDE_51230369906795
// MI455X (gfx1250) — matrix operand order repaired and checked by reading, not compiled, not run on hardware
//
#include <hip/hip_runtime.h>
#include <stdint.h>

// ---------------------------------------------------------------------------
// Problem constants (B,K,M,H,D) = (128, 64, 8192, 600, 200)
// ---------------------------------------------------------------------------
#define B_   128
#define K_   64
#define M_   8192
#define H_   600
#define D_   200
#define HP   608     // H padded to 19*32 (WMMA K chunks)
#define DP   224     // D padded to 7*32  (WMMA K chunks for final GEMM)
#define NP   208     // D padded to 13*16 (WMMA N tiles in gemm2)
#define TAU_INV   10.0f
#define LOG2E     1.4426950408889634f
#define LN2       0.6931471805599453f
#define EPSR      1e-12f

typedef __bf16 bf16_t;
typedef __attribute__((ext_vector_type(16))) bf16_t v16bf;
typedef __attribute__((ext_vector_type(8)))  float  v8f;

static __device__ __forceinline__ unsigned short f2bf(float f) {
  union { float f; unsigned int u; } c; c.f = f;
  unsigned int u = c.u;
  return (unsigned short)((u + 0x7FFFu + ((u >> 16) & 1u)) >> 16);
}

static __device__ __forceinline__ v8f v8zero() {
  v8f v;
  #pragma unroll
  for (int i = 0; i < 8; ++i) v[i] = 0.0f;
  return v;
}

// branch-free tanh: clamp then (e^2x - 1)/(e^2x + 1) via v_exp_f32 + v_rcp_f32
static __device__ __forceinline__ float fast_tanh(float x) {
  float xc = fminf(fmaxf(x, -8.0f), 8.0f);
  float t  = __builtin_amdgcn_exp2f(xc * (2.0f * LOG2E));
  return (t - 1.0f) * __builtin_amdgcn_rcpf(t + 1.0f);
}

// A fragment, 16x32 bf16, row-major source, ld mult of 8, k0 mult of 32.
// Lane l: row = row0 + (l&15); elems 0..15 = K[k0 + (l>>4)*16 ..]
// (two 16B loads).
static __device__ __forceinline__ v16bf load_a(const unsigned short* base, int ld,
                                               int row0, int k0, int lane) {
  int half = lane >> 4, r = lane & 15;
  const unsigned short* p = base + (size_t)(row0 + r) * ld + k0 + half * 16;
  v16bf out;
  ((uint4*)&out)[0] = *(const uint4*)(p);
  ((uint4*)&out)[1] = *(const uint4*)(p + 8);
  return out;
}

// B fragment, 32x16 bf16, from "B transposed" row-major (row n = K-run of col n).
// Lane l: col = col0 + (l&15); lanes 0-15 K=k0..k0+15, lanes 16-31 K=k0+16..k0+31.
static __device__ __forceinline__ v16bf load_b(const unsigned short* base, int ld,
                                               int col0, int k0, int lane) {
  int half = lane >> 4, c = lane & 15;
  const unsigned short* p = base + (size_t)(col0 + c) * ld + k0 + half * 16;
  v16bf out;
  ((uint4*)&out)[0] = ((const uint4*)p)[0];
  ((uint4*)&out)[1] = ((const uint4*)p)[1];
  return out;
}

static __device__ __forceinline__ v8f wmma_bf16(v16bf a, v16bf b, v8f c) {
  return __builtin_amdgcn_wmma_f32_16x16x32_bf16(false, a, false, b, (short)0, c,
                                                 false, false);
}

// ---------------------------------------------------------------------------
// K1a: per-batch gram gather + L1-then-L2 row normalization -> bf16 vals[B][64][64]
// ---------------------------------------------------------------------------
__global__ void k_vals(const float* __restrict__ gram, const int* __restrict__ idx,
                       unsigned short* __restrict__ vals) {
  __shared__ int   sidx[K_];
  __shared__ float g[K_ * K_];
  __shared__ float scale[K_];
  int b = blockIdx.x, t = threadIdx.x;
  if (t < K_) sidx[t] = idx[b * K_ + t];
  __syncthreads();
  for (int e = t; e < K_ * K_; e += 256) {
    int k = e >> 6, j = e & 63;
    g[e] = gram[(size_t)sidx[k] * M_ + sidx[j]];
  }
  __syncthreads();
  if (t < K_) {
    float s1 = 0.0f;
    for (int j = 0; j < K_; ++j) s1 += fabsf(g[t * K_ + j]);
    s1 = fmaxf(s1, EPSR);
    float r1 = 1.0f / s1;
    float s2 = 0.0f;
    for (int j = 0; j < K_; ++j) { float v = g[t * K_ + j] * r1; s2 += v * v; }
    float n2 = fmaxf(sqrtf(s2), EPSR);
    scale[t] = r1 / n2;
  }
  __syncthreads();
  for (int e = t; e < K_ * K_; e += 256)
    vals[(size_t)b * (K_ * K_) + e] = f2bf(g[e] * scale[e >> 6]);
}

// ---------------------------------------------------------------------------
// K1b: row-L2-normalize items -> bf16 Ipad[8192][224] (cols 200..223 zero)
// ---------------------------------------------------------------------------
__global__ void k_items(const float* __restrict__ items, unsigned short* __restrict__ Ipad) {
  int lane = threadIdx.x & 31, w = threadIdx.x >> 5;
  int m = blockIdx.x * 8 + w;
  const float* row = items + (size_t)m * D_;
  float s = 0.0f;
  for (int n = lane; n < D_; n += 32) { float v = row[n]; s += v * v; }
  #pragma unroll
  for (int off = 16; off >= 1; off >>= 1) s += __shfl_xor(s, off, 32);
  float inv = 1.0f / fmaxf(sqrtf(s), EPSR);
  unsigned short* dst = Ipad + (size_t)m * DP;
  for (int n = lane; n < DP; n += 32)
    dst[n] = f2bf(n < D_ ? row[n] * inv : 0.0f);
}

// ---------------------------------------------------------------------------
// K1c: W2T[n][k] = W2[k][n] for n<200,k<600 else 0  (bf16, 208x608)
// ---------------------------------------------------------------------------
__global__ void k_w2t(const float* __restrict__ W2, unsigned short* __restrict__ W2T) {
  int e = blockIdx.x * 256 + threadIdx.x;
  if (e >= NP * HP) return;
  int n = e / HP, k = e - n * HP;
  float v = (n < D_ && k < H_) ? W2[(size_t)k * (2 * D_) + n] : 0.0f;
  W2T[e] = f2bf(v);
}

// ---------------------------------------------------------------------------
// K2: per batch  h = tanh(vals @ W1[idx] + b1)  ->  bf16 h[8192][608]
//     operands + bias staged in LDS; branch-free fast_tanh epilogue
// ---------------------------------------------------------------------------
__global__ void k_gemm1(const unsigned short* __restrict__ vals,
                        const float* __restrict__ W1, const float* __restrict__ b1,
                        const int* __restrict__ idx, unsigned short* __restrict__ h) {
  extern __shared__ __align__(16) unsigned short smem[];
  unsigned short* av  = smem;             // 64*64 bf16
  unsigned short* w1t = smem + K_ * K_;   // [HP][64] bf16 (transposed)
  __shared__ int   sidx[K_];
  __shared__ float sb1[HP];
  int b = blockIdx.x, t = threadIdx.x;
  if (t < K_) sidx[t] = idx[b * K_ + t];
  for (int e = t; e < HP; e += 256) sb1[e] = (e < H_) ? b1[e] : 0.0f;
  const unsigned short* vsrc = vals + (size_t)b * (K_ * K_);
  for (int e = t; e < K_ * K_; e += 256) av[e] = vsrc[e];
  __syncthreads();
  for (int e = t; e < K_ * HP; e += 256) {
    int k = e / HP, n = e - k * HP;
    float v = (n < H_) ? W1[(size_t)sidx[k] * H_ + n] : 0.0f;
    w1t[n * K_ + k] = f2bf(v);
  }
  __syncthreads();
  int lane = t & 31, w = t >> 5;
  int half = lane >> 4, c = lane & 15;
  for (int tile = w; tile < 4 * 38; tile += 8) {
    int rt = tile & 3, ct = tile >> 2;
    int row0 = rt * 16, col0 = ct * 16;
    v8f acc = v8zero();
    #pragma unroll
    for (int k0 = 0; k0 < K_; k0 += 32) {
      v16bf a  = load_a(av,  K_, row0, k0, lane);
      v16bf bb = load_b(w1t, K_, col0, k0, lane);
      acc = wmma_bf16(a, bb, acc);
    }
    int N = col0 + c;
    float bn = sb1[N];
    unsigned short* dst = h + (size_t)(b * K_) * HP;
    #pragma unroll
    for (int r = 0; r < 8; ++r) {
      int Mr = row0 + r + half * 8;
      dst[(size_t)Mr * HP + N] = f2bf(fast_tanh(acc[r] + bn));
    }
  }
}

// ---------------------------------------------------------------------------
// K3: x = h @ W2T' + b2 ; Z = l2norm_rows(x)  ->  bf16 Zpad[8192][224]
// ---------------------------------------------------------------------------
__global__ void k_gemm2(const unsigned short* __restrict__ h,
                        const unsigned short* __restrict__ W2T,
                        const float* __restrict__ b2, unsigned short* __restrict__ Zpad) {
  extern __shared__ __align__(16) float smemf[];
  float* xbuf = smemf;            // [64][NP]
  float* rinv = smemf + 64 * NP;  // [64]
  __shared__ float sb2[NP];
  int t = threadIdx.x, lane = t & 31, w = t >> 5;
  if (t < NP) sb2[t] = (t < D_) ? b2[t] : 0.0f;
  int rowBlock = blockIdx.x * 64;
  int rt = w & 3, hfl = w >> 2;
  int ct0 = hfl ? 7 : 0, ctn = hfl ? 6 : 7;
  v8f acc[7];
  #pragma unroll
  for (int i = 0; i < 7; ++i) acc[i] = v8zero();
  for (int k0 = 0; k0 < HP; k0 += 32) {
    v16bf a = load_a(h, HP, rowBlock + rt * 16, k0, lane);
    for (int i = 0; i < ctn; ++i) {
      v16bf bb = load_b(W2T, HP, (ct0 + i) * 16, k0, lane);
      acc[i] = wmma_bf16(a, bb, acc[i]);
    }
  }
  __syncthreads();  // sb2 ready; also all waves done writing nothing yet
  int half = lane >> 4, c = lane & 15;
  for (int i = 0; i < ctn; ++i) {
    int N = (ct0 + i) * 16 + c;
    float bn = sb2[N];
    #pragma unroll
    for (int r = 0; r < 8; ++r)
      xbuf[(rt * 16 + r + half * 8) * NP + N] = acc[i][r] + bn;
  }
  __syncthreads();
  if (t < 64) {
    float s = 0.0f;
    for (int n = 0; n < D_; ++n) { float v = xbuf[t * NP + n]; s += v * v; }
    rinv[t] = 1.0f / fmaxf(sqrtf(s), EPSR);
  }
  __syncthreads();
  for (int e = t; e < 64 * DP; e += 256) {
    int r = e / DP, n = e - r * DP;
    float v = (n < D_) ? xbuf[r * NP + n] * rinv[r] : 0.0f;
    Zpad[(size_t)(rowBlock + r) * DP + n] = f2bf(v);
  }
}

// ---------------------------------------------------------------------------
// K4: out[b,m] = log1p( mean_k exp( (Z[b*64+k,:] . I[m,:]) / tau ) )
//     Z batch staged into LDS via gfx1250 async-to-LDS DMA (ASYNCcnt),
//     items panel prefetched (global_prefetch_b8). 7 WMMA K-steps, fused
//     exp/mean/log1p epilogue with a lane-16 xor-shuffle reduction.
// ---------------------------------------------------------------------------
__global__ void k_score(const unsigned short* __restrict__ Zpad,
                        const unsigned short* __restrict__ Ipad,
                        float* __restrict__ out) {
  extern __shared__ __align__(16) unsigned short zb[]; // [64][DP] bf16
  int t = threadIdx.x, lane = t & 31, w = t >> 5;
  int b = blockIdx.y;
  int wcol = blockIdx.x * 256 + w * 32;
  const unsigned short* zsrc = Zpad + (size_t)b * K_ * DP;

  // async copy Z batch (64 x 224 bf16 = 28KB) to LDS: 7 x b128 per thread
  for (int e = t * 8; e < K_ * DP; e += 256 * 8) {
    unsigned ldsoff = (unsigned)(unsigned long long)(const void*)(zb + e);
    unsigned long long ga = (unsigned long long)(const void*)(zsrc + e);
    asm volatile("global_load_async_to_lds_b128 %0, %1, off"
                 :: "v"(ldsoff), "v"(ga) : "memory");
  }
  // prefetch this wave's items panel (per-lane address -> 32 rows covered)
  __builtin_prefetch(Ipad + (size_t)(wcol + (lane & 15) + (lane >> 4) * 16) * DP, 0, 0);
  asm volatile("s_wait_asynccnt 0x0" ::: "memory");
  __syncthreads();

  v8f acc[4][2];
  #pragma unroll
  for (int rt = 0; rt < 4; ++rt)
    #pragma unroll
    for (int ct = 0; ct < 2; ++ct) acc[rt][ct] = v8zero();
  #pragma unroll
  for (int k0 = 0; k0 < DP; k0 += 32) {
    v16bf a0 = load_a(zb,  DP,  0, k0, lane);
    v16bf a1 = load_a(zb,  DP, 16, k0, lane);
    v16bf a2 = load_a(zb,  DP, 32, k0, lane);
    v16bf a3 = load_a(zb,  DP, 48, k0, lane);
    v16bf b0 = load_b(Ipad, DP, wcol,      k0, lane);
    v16bf b1 = load_b(Ipad, DP, wcol + 16, k0, lane);
    acc[0][0] = wmma_bf16(a0, b0, acc[0][0]);
    acc[1][0] = wmma_bf16(a1, b0, acc[1][0]);
    acc[2][0] = wmma_bf16(a2, b0, acc[2][0]);
    acc[3][0] = wmma_bf16(a3, b0, acc[3][0]);
    acc[0][1] = wmma_bf16(a0, b1, acc[0][1]);
    acc[1][1] = wmma_bf16(a1, b1, acc[1][1]);
    acc[2][1] = wmma_bf16(a2, b1, acc[2][1]);
    acc[3][1] = wmma_bf16(a3, b1, acc[3][1]);
  }
  // exp + sum over the 64 k-rows: lane holds 32 rows across 4 frags; lanes
  // l and l^16 hold complementary halves of the same column.
  #pragma unroll
  for (int ct = 0; ct < 2; ++ct) {
    float s = 0.0f;
    #pragma unroll
    for (int rt = 0; rt < 4; ++rt)
      #pragma unroll
      for (int r = 0; r < 8; ++r)
        s += __builtin_amdgcn_exp2f(acc[rt][ct][r] * (TAU_INV * LOG2E));
    s += __shfl_xor(s, 16, 32);
    if (lane < 16) {
      int col = wcol + ct * 16 + lane;
      out[(size_t)b * M_ + col] =
          __builtin_amdgcn_logf(s * (1.0f / K_) + 1.0f) * LN2;
    }
  }
}

// ---------------------------------------------------------------------------
// Host-side launcher. Inputs (setup_inputs order):
//  0 rating(f32, unused) 1 gram 2 W1 3 b1 4 W2 5 b2 6 items 7 item_idx(i32)
// ---------------------------------------------------------------------------
extern "C" void kernel_launch(void* const* d_in, const int* in_sizes, int n_in,
                              void* d_out, int out_size, void* d_ws, size_t ws_size,
                              hipStream_t stream) {
  const float* gram  = (const float*)d_in[1];
  const float* W1    = (const float*)d_in[2];
  const float* b1    = (const float*)d_in[3];
  const float* W2    = (const float*)d_in[4];
  const float* b2    = (const float*)d_in[5];
  const float* items = (const float*)d_in[6];
  const int*   idx   = (const int*)d_in[7];
  float* out = (float*)d_out;

  char* ws = (char*)d_ws;
  unsigned short* vals = (unsigned short*)(ws + 0);         // 128*64*64   -> 1,048,576 B
  unsigned short* Ipad = (unsigned short*)(ws + 1048576);   // 8192*224    -> 3,670,016 B
  unsigned short* hbuf = (unsigned short*)(ws + 4718592);   // 8192*608    -> 9,961,472 B
  unsigned short* W2T  = (unsigned short*)(ws + 14680064);  // 208*608     ->   252,928 B
  unsigned short* Zpad = (unsigned short*)(ws + 14932992);  // 8192*224    -> 3,670,016 B
  // total workspace: 18,603,008 bytes

  k_vals <<<B_, 256, 0, stream>>>(gram, idx, vals);
  k_items<<<M_ / 8, 256, 0, stream>>>(items, Ipad);
  k_w2t  <<<(NP * HP + 255) / 256, 256, 0, stream>>>(W2, W2T);
  k_gemm1<<<B_, 256, (K_ * K_ + K_ * HP) * 2, stream>>>(vals, W1, b1, idx, hbuf);
  k_gemm2<<<M_ / 64, 256, (64 * NP + 64) * 4, stream>>>(hbuf, W2T, b2, Zpad);
  k_score<<<dim3(M_ / 256, B_), 256, K_ * DP * 2, stream>>>(Zpad, Ipad, out);
}
